// InteractionNetSingleTagger_56977036149125
// MI455X (gfx1250) — matrix operand order repaired and checked by reading, not compiled, not run on hardware
//
#include <hip/hip_runtime.h>

typedef __attribute__((ext_vector_type(16))) __bf16 v16bf;
typedef __attribute__((ext_vector_type(8)))  __bf16 v8bf;
typedef __attribute__((ext_vector_type(8)))  float  v8f;
typedef __attribute__((ext_vector_type(4)))  unsigned int v4u;

#define B_   256
#define P_   32
#define N_   60
#define NR   3540   // 60*59 directed edges
#define EPSV 1e-5f

union BFr { v16bf v; v8bf h[2]; };

__device__ __forceinline__ v8f wmma_bf16(v16bf a, v16bf b, v8f c) {
  // D = A(16x32) * B(32x16) + C, fp32 accum
  return __builtin_amdgcn_wmma_f32_16x16x32_bf16(false, a, false, b, (short)0, c, false, false);
}

// ---------------- utility kernels ----------------

__global__ void k_zero(float* p, int n) {
  int i = blockIdx.x * blockDim.x + threadIdx.x;
  if (i < n) p[i] = 0.f;
}

__global__ void k_cvt_bf16(const float* s, __bf16* d, int n) {
  int i = blockIdx.x * blockDim.x + threadIdx.x;
  if (i < n) d[i] = (__bf16)s[i];
}

// input BN stats: per-P mean/var over (B, N) -> scale/shift
__global__ void k_input_stats(const float* x, const float* g, const float* be,
                              float* s_in, float* t_in) {
  __shared__ float sm[256], sq[256];
  int p = blockIdx.x, tid = threadIdx.x;
  float a = 0.f, b = 0.f;
  for (int t = tid; t < B_ * N_; t += 256) {
    int bb = t / N_, n = t - bb * N_;
    float f = x[(bb * P_ + p) * N_ + n];
    a += f; b += f * f;
  }
  sm[tid] = a; sq[tid] = b; __syncthreads();
  for (int o = 128; o > 0; o >>= 1) {
    if (tid < o) { sm[tid] += sm[tid + o]; sq[tid] += sq[tid + o]; }
    __syncthreads();
  }
  if (tid == 0) {
    float cnt = (float)(B_ * N_);
    float m = sm[0] / cnt, v = sq[0] / cnt - m * m;
    float inv = rsqrtf(v + EPSV);
    float s = g[p] * inv;
    s_in[p] = s; t_in[p] = be[p] - m * s;
  }
}

// xt[b][n][p] = bf16(bn(x[b][p][n]))  (node features, transposed, normalized)
__global__ void k_make_xt(const float* x, const float* s_in, const float* t_in, __bf16* xt) {
  int i = blockIdx.x * 256 + threadIdx.x;
  if (i >= B_ * N_ * P_) return;
  int p = i & 31, bn = i >> 5;
  int n = bn % N_, b = bn / N_;
  xt[i] = (__bf16)(s_in[p] * x[(b * P_ + p) * N_ + n] + t_in[p]);
}

// finalize BN stats: sum/sumsq -> scale/shift
__global__ void k_finalize(const float* sum, const float* sq, float cnt,
                           const float* g, const float* be, float* s, float* t, int nch) {
  int i = threadIdx.x;
  if (i >= nch) return;
  float m = sum[i] / cnt, v = sq[i] / cnt - m * m;
  float inv = rsqrtf(v + EPSV);
  float ss = g[i] * inv;
  s[i] = ss; t[i] = be[i] - m * ss;
}

// ---------------- fr1: edge-gather GEMM (906240 x 64) @ (64 x 64)^T ----------------
// grid (28, B), block 256 (8 waves). Wave tile = 16 edges x 64 out.
__global__ void k_gemm_edges(const __bf16* xt, const __bf16* wb, const float* bias,
                             __bf16* H, float* gsum, float* gsq) {
  __shared__ __align__(16) __bf16 sx[N_ * P_];     // this batch item's nodes (3840 B)
  __shared__ __align__(32) __bf16 wsh[64 * 64];    // fr1 weights, (out,in) row-major
  __shared__ __align__(16) __bf16 hst[128 * 64];   // staged output tile
  __shared__ float lsum[64], lsq[64];
  int tid = threadIdx.x, b = blockIdx.y, ebase = blockIdx.x * 128;
  {
    const v4u* src = (const v4u*)(xt + (size_t)b * N_ * P_);
    v4u* dst = (v4u*)sx;
    for (int c = tid; c < N_ * P_ / 8; c += 256) dst[c] = src[c];
    const v4u* ws = (const v4u*)wb; v4u* wd = (v4u*)wsh;
    for (int c = tid; c < 64 * 8; c += 256) wd[c] = ws[c];
  }
  if (tid < 64) { lsum[tid] = 0.f; lsq[tid] = 0.f; }
  __syncthreads();

  int wid = tid >> 5, lane = tid & 31, nl = lane & 15, kh = lane >> 4, kb = kh * 16;
  int te = ebase + wid * 16;
  int e = te + nl; int ec = e < NR ? e : NR - 1;
  int nrcv = ec / 59; int js = ec - nrcv * 59; int nsnd = js + (js >= nrcv);

  v8f acc[4];
  for (int nt = 0; nt < 4; ++nt) {
    float bv = bias[nt * 16 + nl];
    for (int i = 0; i < 8; ++i) acc[nt][i] = bv;
  }
  for (int kc = 0; kc < 2; ++kc) {            // kc=0: recv half, kc=1: send half
    int node = kc ? nsnd : nrcv;
    BFr a;
    a.h[0] = *(const v8bf*)(sx + node * P_ + kb);
    a.h[1] = *(const v8bf*)(sx + node * P_ + kb + 8);
    for (int nt = 0; nt < 4; ++nt) {
      v16bf bf = *(const v16bf*)(wsh + (nt * 16 + nl) * 64 + kc * 32 + kh * 16);
      acc[nt] = wmma_bf16(a.v, bf, acc[nt]);
    }
  }
  for (int nt = 0; nt < 4; ++nt) {
    int ch = nt * 16 + nl; float sm = 0.f, s2 = 0.f;
    for (int i = 0; i < 8; ++i) {
      int r = kh * 8 + i;
      float f = acc[nt][i];
      hst[(wid * 16 + r) * 64 + ch] = (__bf16)f;
      if (te + r < NR) { sm += f; s2 += f * f; }   // mask padded edges from stats
    }
    atomicAdd(&lsum[ch], sm); atomicAdd(&lsq[ch], s2);
  }
  __syncthreads();
  const v4u* src = (const v4u*)hst;
  for (int idx = tid; idx < 128 * 8; idx += 256) {
    int r = idx >> 3, c = idx & 7, ee = ebase + r;
    if (ee < NR) ((v4u*)(H + ((size_t)b * NR + ee) * 64))[c] = src[idx];
  }
  for (int ch = tid; ch < 64; ch += 256) {
    atomicAdd(&gsum[ch], lsum[ch]); atomicAdd(&gsq[ch], lsq[ch]);
  }
}

// ---------------- generic GEMM: out = (opt bn+relu)(A) @ W^T + bias; accumulate stats ----------------
// A: Mtot x 64 bf16 (Mtot multiple of 128), W: (NOUT x 64) bf16 row-major.
template <int NOUT, bool BN>
__global__ void k_gemm(const __bf16* A, const __bf16* W, const float* bias,
                       const float* s_in, const float* t_in,
                       __bf16* H, float* gsum, float* gsq) {
  __shared__ __align__(32) __bf16 wsh[NOUT * 64];
  __shared__ __align__(16) __bf16 hst[128 * NOUT];
  __shared__ float lsum[NOUT], lsq[NOUT];
  __shared__ float ssh[64], tsh[64];
  int tid = threadIdx.x;
  {
    const v4u* ws = (const v4u*)W; v4u* wd = (v4u*)wsh;
    for (int c = tid; c < NOUT * 8; c += 256) wd[c] = ws[c];
  }
  if (BN && tid < 64) { ssh[tid] = s_in[tid]; tsh[tid] = t_in[tid]; }
  if (tid < NOUT) { lsum[tid] = 0.f; lsq[tid] = 0.f; }
  __syncthreads();

  int wid = tid >> 5, lane = tid & 31, nl = lane & 15, kh = lane >> 4, kb = kh * 16;
  size_t row = (size_t)blockIdx.x * 128 + wid * 16 + nl;
  const __bf16* ap = A + row * 64;
  constexpr int NT = NOUT / 16;
  v8f acc[NT];
  for (int nt = 0; nt < NT; ++nt) {
    float bv = bias[nt * 16 + nl];
    for (int i = 0; i < 8; ++i) acc[nt][i] = bv;
  }
  for (int kc = 0; kc < 64; kc += 32) {
    BFr a;
    a.h[0] = *(const v8bf*)(ap + kc + kb);
    a.h[1] = *(const v8bf*)(ap + kc + kb + 8);
    if (BN) {
      for (int j = 0; j < 8; ++j) {
        int c0 = kc + kb + j, c1 = kc + kb + 8 + j;
        float f0 = ssh[c0] * (float)a.h[0][j] + tsh[c0];
        float f1 = ssh[c1] * (float)a.h[1][j] + tsh[c1];
        a.h[0][j] = (__bf16)fmaxf(f0, 0.f);
        a.h[1][j] = (__bf16)fmaxf(f1, 0.f);
      }
    }
    for (int nt = 0; nt < NT; ++nt) {
      v16bf bf = *(const v16bf*)(wsh + (nt * 16 + nl) * 64 + kc + kh * 16);
      acc[nt] = wmma_bf16(a.v, bf, acc[nt]);
    }
  }
  for (int nt = 0; nt < NT; ++nt) {
    int ch = nt * 16 + nl; float sm = 0.f, s2 = 0.f;
    for (int i = 0; i < 8; ++i) {
      float f = acc[nt][i];
      hst[(wid * 16 + kh * 8 + i) * NOUT + ch] = (__bf16)f;
      sm += f; s2 += f * f;
    }
    atomicAdd(&lsum[ch], sm); atomicAdd(&lsq[ch], s2);
  }
  __syncthreads();
  {
    const v4u* src = (const v4u*)hst;
    v4u* dst = (v4u*)(H + (size_t)blockIdx.x * 128 * NOUT);
    for (int c = tid; c < 16 * NOUT; c += 256) dst[c] = src[c];
  }
  for (int ch = tid; ch < NOUT; ch += 256) {
    atomicAdd(&gsum[ch], lsum[ch]); atomicAdd(&gsq[ch], lsq[ch]);
  }
}

// ---------------- Ebar reduction (+fr3 bn/relu) and concat into fo-input C ----------------
// recv(e) = e/59 -> edges for node n are contiguous [n*59, n*59+59)
__global__ void k_ebar_concat(const __bf16* E3, const __bf16* xt,
                              const float* s3, const float* t3, __bf16* C) {
  int tid = threadIdx.x, sub = tid >> 5, ch = tid & 31;
  int row = blockIdx.x * 8 + sub;            // (b*60+n)
  int b = row / N_, n = row - b * N_;
  float s = s3[ch], t = t3[ch], acc = 0.f;
  const __bf16* ep = E3 + ((size_t)b * NR + n * 59) * 32 + ch;
  for (int k = 0; k < 59; ++k) {
    float f = s * (float)ep[k * 32] + t;
    acc += fmaxf(f, 0.f);
  }
  C[(size_t)row * 64 + ch]      = xt[(size_t)row * 32 + ch];
  C[(size_t)row * 64 + 32 + ch] = (__bf16)acc;
}

// ---------------- pool over N + final FC ----------------
__global__ void k_pool_fc(const __bf16* O3, const float* s6, const float* t6,
                          const float* fcw, const float* fcb, float* out) {
  __shared__ float pooled[32];
  int b = blockIdx.x, tid = threadIdx.x;
  if (tid < 32) {
    float s = s6[tid], t = t6[tid], acc = 0.f;
    const __bf16* op = O3 + (size_t)b * N_ * 32 + tid;
    for (int n = 0; n < N_; ++n) {
      float f = s * (float)op[n * 32] + t;
      acc += fmaxf(f, 0.f);
    }
    pooled[tid] = acc;
  }
  __syncthreads();
  if (tid < 2) {
    float a = fcb[tid];
    for (int d = 0; d < 32; ++d) a += pooled[d] * fcw[tid * 32 + d];
    out[b * 2 + tid] = a;
  }
}

extern "C" void kernel_launch(void* const* d_in, const int* in_sizes, int n_in,
                              void* d_out, int out_size, void* d_ws, size_t ws_size,
                              hipStream_t stream) {
  (void)in_sizes; (void)n_in; (void)out_size; (void)ws_size;
  const float* x     = (const float*)d_in[0];
  const float* bnx_g = (const float*)d_in[1];
  const float* bnx_b = (const float*)d_in[2];
  const float* W[6]  = { (const float*)d_in[3],  (const float*)d_in[7],
                         (const float*)d_in[11], (const float*)d_in[15],
                         (const float*)d_in[19], (const float*)d_in[23] };
  const float* Bi[6] = { (const float*)d_in[4],  (const float*)d_in[8],
                         (const float*)d_in[12], (const float*)d_in[16],
                         (const float*)d_in[20], (const float*)d_in[24] };
  const float* Ga[6] = { (const float*)d_in[5],  (const float*)d_in[9],
                         (const float*)d_in[13], (const float*)d_in[17],
                         (const float*)d_in[21], (const float*)d_in[25] };
  const float* Be[6] = { (const float*)d_in[6],  (const float*)d_in[10],
                         (const float*)d_in[14], (const float*)d_in[18],
                         (const float*)d_in[22], (const float*)d_in[26] };
  const float* fcw = (const float*)d_in[27];
  const float* fcb = (const float*)d_in[28];
  float* out = (float*)d_out;

  char* w = (char*)d_ws;
  float* stats = (float*)w;
  float* s_in = stats, *t_in = stats + 32;
  float *LS[6], *LQ[6], *Ls[6], *Lt[6];
  for (int L = 0; L < 6; ++L) {
    float* base = stats + 64 + L * 256;
    LS[L] = base; LQ[L] = base + 64; Ls[L] = base + 128; Lt[L] = base + 192;
  }
  size_t off = 8192;
  auto take = [&](size_t bytes) { void* p = w + off; off += (bytes + 255) & ~(size_t)255; return p; };
  __bf16* wb[6];
  const int wn[6] = { 4096, 4096, 2048, 4096, 4096, 2048 };
  for (int L = 0; L < 6; ++L) wb[L] = (__bf16*)take((size_t)wn[L] * 2);
  __bf16* xt = (__bf16*)take((size_t)B_ * N_ * P_ * 2);
  __bf16* H1 = (__bf16*)take((size_t)B_ * NR * 64 * 2);
  __bf16* H2 = (__bf16*)take((size_t)B_ * NR * 64 * 2);
  __bf16* E3 = (__bf16*)take((size_t)B_ * NR * 32 * 2);
  __bf16* Cm = (__bf16*)take((size_t)B_ * N_ * 64 * 2);
  __bf16* G1 = (__bf16*)take((size_t)B_ * N_ * 64 * 2);
  __bf16* G2 = (__bf16*)take((size_t)B_ * N_ * 64 * 2);
  __bf16* O3 = (__bf16*)take((size_t)B_ * N_ * 32 * 2);

  const float edgeCnt = (float)B_ * NR;   // 906240
  const float objCnt  = (float)B_ * N_;   // 15360

  k_zero<<<7, 256, 0, stream>>>(stats, 1600);
  k_input_stats<<<32, 256, 0, stream>>>(x, bnx_g, bnx_b, s_in, t_in);
  k_make_xt<<<(B_ * N_ * P_) / 256, 256, 0, stream>>>(x, s_in, t_in, xt);
  for (int L = 0; L < 6; ++L)
    k_cvt_bf16<<<(wn[L] + 255) / 256, 256, 0, stream>>>(W[L], wb[L], wn[L]);

  // fr pipeline over 906240 edge rows
  k_gemm_edges<<<dim3(28, B_), 256, 0, stream>>>(xt, wb[0], Bi[0], H1, LS[0], LQ[0]);
  k_finalize<<<1, 64, 0, stream>>>(LS[0], LQ[0], edgeCnt, Ga[0], Be[0], Ls[0], Lt[0], 64);
  k_gemm<64, true><<<7080, 256, 0, stream>>>(H1, wb[1], Bi[1], Ls[0], Lt[0], H2, LS[1], LQ[1]);
  k_finalize<<<1, 64, 0, stream>>>(LS[1], LQ[1], edgeCnt, Ga[1], Be[1], Ls[1], Lt[1], 64);
  k_gemm<32, true><<<7080, 256, 0, stream>>>(H2, wb[2], Bi[2], Ls[1], Lt[1], E3, LS[2], LQ[2]);
  k_finalize<<<1, 64, 0, stream>>>(LS[2], LQ[2], edgeCnt, Ga[2], Be[2], Ls[2], Lt[2], 32);

  // aggregate edges -> nodes, build fo input
  k_ebar_concat<<<(B_ * N_) / 8, 256, 0, stream>>>(E3, xt, Ls[2], Lt[2], Cm);

  // fo pipeline over 15360 node rows
  k_gemm<64, false><<<120, 256, 0, stream>>>(Cm, wb[3], Bi[3], nullptr, nullptr, G1, LS[3], LQ[3]);
  k_finalize<<<1, 64, 0, stream>>>(LS[3], LQ[3], objCnt, Ga[3], Be[3], Ls[3], Lt[3], 64);
  k_gemm<64, true><<<120, 256, 0, stream>>>(G1, wb[4], Bi[4], Ls[3], Lt[3], G2, LS[4], LQ[4]);
  k_finalize<<<1, 64, 0, stream>>>(LS[4], LQ[4], objCnt, Ga[4], Be[4], Ls[4], Lt[4], 64);
  k_gemm<32, true><<<120, 256, 0, stream>>>(G2, wb[5], Bi[5], Ls[4], Lt[4], O3, LS[5], LQ[5]);
  k_finalize<<<1, 64, 0, stream>>>(LS[5], LQ[5], objCnt, Ga[5], Be[5], Ls[5], Lt[5], 32);

  // pool + final classifier
  k_pool_fc<<<B_, 32, 0, stream>>>(O3, Ls[5], Lt[5], fcw, fcb, out);
}
